// HSTULiteDeep_84232898609282
// MI455X (gfx1250) — matrix operand order repaired and checked by reading, not compiled, not run on hardware
//
#include <hip/hip_runtime.h>
#include <hip/hip_bf16.h>

// ---------------- model constants (match reference) ----------------
#define BATCH_N   4096
#define SEQ_T     39
#define D_MODEL_N 256
#define N_HEADS_N 8
#define DH_N      32
#define D_FF_N    1024
#define N_LAYERS_N 8
#define VOCABP1   10001
#define N_SPARSE_N 26
#define N_DENSE_N  13
#define M_ROWS   (BATCH_N * SEQ_T)          // 159744 (divisible by 128)

typedef __attribute__((ext_vector_type(16))) __bf16           v16bf;
typedef __attribute__((ext_vector_type(16))) unsigned short   v16u;
typedef __attribute__((ext_vector_type(8)))  float            v8f;

// ---------------- helpers ----------------
__device__ __forceinline__ unsigned short f32_to_bf16(float f) {
  unsigned int u = __float_as_uint(f);
  if ((u & 0x7fffffffu) > 0x7f800000u) return (unsigned short)((u >> 16) | 0x40u); // quiet NaN
  u += 0x7fffu + ((u >> 16) & 1u);   // round-to-nearest-even
  return (unsigned short)(u >> 16);
}
__device__ __forceinline__ float bf16_to_f32(unsigned short h) {
  return __uint_as_float(((unsigned int)h) << 16);
}
__device__ __forceinline__ float wave_sum(float v) {
  #pragma unroll
  for (int s = 16; s > 0; s >>= 1) v += __shfl_xor(v, s, 32);
  return v;
}

// ---------------- fused f32 -> bf16 convert + transpose: src[N,K] -> dst[K,N] -------------
__global__ void __launch_bounds__(256)
cvt_transpose_bf16_kernel(const float* __restrict__ src, unsigned short* __restrict__ dst,
                          int N, int K) {
  size_t i = (size_t)blockIdx.x * 256 + threadIdx.x;    // over N*K
  if (i >= (size_t)N * K) return;
  int n = (int)(i / K);
  int k = (int)(i - (size_t)n * K);
  dst[(size_t)k * N + n] = f32_to_bf16(src[i]);
}

// ---------------- embedding (dense projection + sparse gather) ----------------
__global__ void __launch_bounds__(256)
embed_kernel(const float* __restrict__ dense_x, const int* __restrict__ sparse_x,
             const float* __restrict__ Wd, const float* __restrict__ bd,
             const float* __restrict__ emb,
             float* __restrict__ x, float* __restrict__ x0) {
  size_t idx = (size_t)blockIdx.x * 256 + threadIdx.x;   // over B*39*256
  int d = (int)(idx & 255);
  int t = (int)((idx >> 8) % SEQ_T);
  int b = (int)(idx / ((size_t)SEQ_T * D_MODEL_N));
  float v;
  if (t < N_DENSE_N) {
    int orow = t * D_MODEL_N + d;                 // row of W_dense [3328,13]
    const float* wr = Wd + (size_t)orow * N_DENSE_N;
    const float* xr = dense_x + (size_t)b * N_DENSE_N;
    v = bd[orow];
    #pragma unroll
    for (int j = 0; j < N_DENSE_N; ++j) v += xr[j] * wr[j];
  } else {
    int s = t - N_DENSE_N;
    int id = sparse_x[(size_t)b * N_SPARSE_N + s];
    v = (id == 0) ? 0.0f : emb[(((size_t)s * VOCABP1 + id) * D_MODEL_N) + d];
  }
  x[idx] = v;
  x0[idx] = v;
}

// ---------------- LayerNorm (fp32 in -> bf16 out), one wave per row of 256 ----------------
__global__ void __launch_bounds__(256)
layernorm_bf16_kernel(const float* __restrict__ x, const float* __restrict__ g,
                      const float* __restrict__ bta, unsigned short* __restrict__ out, int M) {
  int wid = blockIdx.x * 8 + (threadIdx.x >> 5);
  int lane = threadIdx.x & 31;
  if (wid >= M) return;
  const float* row = x + (size_t)wid * D_MODEL_N;
  float v[8]; float s = 0.f;
  #pragma unroll
  for (int j = 0; j < 8; ++j) { v[j] = row[lane + 32 * j]; s += v[j]; }
  s = wave_sum(s);
  float mean = s * (1.0f / D_MODEL_N);
  float var = 0.f;
  #pragma unroll
  for (int j = 0; j < 8; ++j) { float d = v[j] - mean; var += d * d; }
  var = wave_sum(var) * (1.0f / D_MODEL_N);
  float inv = rsqrtf(var + 1e-5f);
  unsigned short* orow = out + (size_t)wid * D_MODEL_N;
  #pragma unroll
  for (int j = 0; j < 8; ++j) {
    int c = lane + 32 * j;
    orow[c] = f32_to_bf16((v[j] - mean) * inv * g[c] + bta[c]);
  }
}

// ---------------- WMMA bf16 GEMM:  C[M,N] = act(A[M,K] * Wt[K,N] + bias) (+ res) ----------
// Wt is the PRE-TRANSPOSED weight ([K,N] row-major, ld = ldW).
// 128x128 block tile, 256 threads = 8 waves, each wave 32x64 (2x4 WMMA tiles), K step 32.
// Register-buffered global->LDS pipeline + distance-2 prefetch. M,N % 128 == 0; K % 32 == 0.
template<bool RELU, bool HAS_RES, bool STORE_F32, bool STORE_BF16>
__global__ void __launch_bounds__(256)
gemm_wmma_bf16(const unsigned short* __restrict__ A, int ldA,
               const unsigned short* __restrict__ Wt, int ldW, int K,
               const float* __restrict__ bias,
               float* __restrict__ C, int ldC,
               unsigned short* __restrict__ Cb, int ldCb,
               const float* __restrict__ Res, int ldRes) {
  __shared__ unsigned short As[128 * 32];    // [m][k], row stride 32 (64B)
  __shared__ unsigned short Bs[32 * 136];    // [k][n], row stride 136 (272B, 16B-aligned, bank-spread)
  const int tid  = threadIdx.x;
  const int lane = tid & 31;
  const int wave = tid >> 5;
  const int m0 = blockIdx.x * 128;
  const int n0 = blockIdx.y * 128;
  const int wm = (wave >> 1) * 32;           // 0,32,64,96
  const int wn = (wave & 1) * 64;            // 0,64

  // A staging: thread -> (row = tid>>1 in 0..127, half = 16*(tid&1))
  const int arow  = tid >> 1;
  const int ahalf = (tid & 1) * 16;
  const unsigned short* aSrc = A + (size_t)(m0 + arow) * ldA + ahalf;
  unsigned short* aDst = As + arow * 32 + ahalf;
  // B staging: thread -> (k = tid>>3 in 0..31, seg = 16*(tid&7))
  const int bk   = tid >> 3;
  const int bseg = (tid & 7) * 16;
  const unsigned short* bSrc = Wt + (size_t)bk * ldW + n0 + bseg;
  unsigned short* bDst = Bs + bk * 136 + bseg;

  v8f acc[2][4] = {};
  uint4 ra0, ra1, rb0, rb1;
  ra0 = *(const uint4*)(aSrc);
  ra1 = *(const uint4*)(aSrc + 8);
  rb0 = *(const uint4*)(bSrc);
  rb1 = *(const uint4*)(bSrc + 8);

  for (int k0 = 0; k0 < K; k0 += 32) {
    // commit staged registers to LDS
    *(uint4*)(aDst)     = ra0;
    *(uint4*)(aDst + 8) = ra1;
    *(uint4*)(bDst)     = rb0;
    *(uint4*)(bDst + 8) = rb1;
    __syncthreads();
    // start next tile's global loads (overlap with WMMA below)
    if (k0 + 32 < K) {
      const unsigned short* an = aSrc + (k0 + 32);
      const unsigned short* bn = bSrc + (size_t)(k0 + 32) * ldW;
      ra0 = *(const uint4*)(an);
      ra1 = *(const uint4*)(an + 8);
      rb0 = *(const uint4*)(bn);
      rb1 = *(const uint4*)(bn + 8);
      if (k0 + 64 < K) {                      // distance-2 prefetch -> global_prefetch_b8
        __builtin_prefetch(aSrc + (k0 + 64), 0, 1);
        __builtin_prefetch(bSrc + (size_t)(k0 + 64) * ldW, 0, 1);
      }
    }
    // B fragments: lane holds N-col = lane&15; lanes 0-15 -> K {0..7,16..23}; lanes 16-31 -> K {8..15,24..31}
    v16u bu[4];
    #pragma unroll
    for (int j = 0; j < 4; ++j) {
      const unsigned short* bp = Bs + wn + 16 * j + (lane & 15);
      const int bkb = (lane < 16) ? 0 : 8;
      #pragma unroll
      for (int s = 0; s < 16; ++s) {
        const int kk = (s < 8) ? (bkb + s) : (16 + bkb + (s - 8));
        bu[j][s] = bp[kk * 136];
      }
    }
    // A fragments: lanes 0-15 -> K {0..7,16..23}; lanes 16-31 -> K {8..15,24..31}
    const int kb = (lane < 16) ? 0 : 8;
    const int mr = lane & 15;
    #pragma unroll
    for (int i = 0; i < 2; ++i) {
      const unsigned short* apf = As + (wm + 16 * i + mr) * 32;
      v16u au;
      *((uint4*)&au)     = *(const uint4*)(apf + kb);
      *((uint4*)&au + 1) = *(const uint4*)(apf + 16 + kb);
      v16bf av = __builtin_bit_cast(v16bf, au);
      #pragma unroll
      for (int j = 0; j < 4; ++j) {
        v16bf bv = __builtin_bit_cast(v16bf, bu[j]);
        acc[i][j] = __builtin_amdgcn_wmma_f32_16x16x32_bf16(
            false, av, false, bv, (short)0, acc[i][j], false, false);
      }
    }
    __syncthreads();
  }

  // epilogue: C/D layout — VGPR r: lanes 0-15 row M=r, lanes 16-31 row M=r+8; col N=lane&15
  #pragma unroll
  for (int i = 0; i < 2; ++i) {
    #pragma unroll
    for (int j = 0; j < 4; ++j) {
      const int col = n0 + wn + 16 * j + (lane & 15);
      const float bcol = bias ? bias[col] : 0.0f;
      #pragma unroll
      for (int r = 0; r < 8; ++r) {
        const int row = m0 + wm + 16 * i + ((lane < 16) ? r : r + 8);
        float v = acc[i][j][r] + bcol;
        if (RELU) v = fmaxf(v, 0.0f);
        if (HAS_RES) v += Res[(size_t)row * ldRes + col];
        if (STORE_F32)  C [(size_t)row * ldC  + col] = v;
        if (STORE_BF16) Cb[(size_t)row * ldCb + col] = f32_to_bf16(v);
      }
    }
  }
}

// ---------------- attention: one wave per (b, h); K/V cached in registers ----------------
// qkv laid out [B*SEQ, 768] with q at col 0, k at 256, v at 512. dh=32 -> lane == channel.
template<bool SOFTMAX, bool OUT_BF16>
__global__ void __launch_bounds__(256)
attention_kernel(const float* __restrict__ qkv, float scale,
                 const float* __restrict__ res,
                 float* __restrict__ outf, unsigned short* __restrict__ outb) {
  const int wid  = blockIdx.x * 8 + (threadIdx.x >> 5);   // [0, B*H)
  const int lane = threadIdx.x & 31;
  const int h = wid & (N_HEADS_N - 1);
  const int b = wid >> 3;
  const size_t base = (size_t)b * SEQ_T * 768;
  const int hc = h * DH_N + lane;

  float kreg[SEQ_T], vreg[SEQ_T];
  #pragma unroll
  for (int k = 0; k < SEQ_T; ++k) {
    kreg[k] = qkv[base + (size_t)k * 768 + 256 + hc];
    vreg[k] = qkv[base + (size_t)k * 768 + 512 + hc];
  }
  #pragma unroll 1
  for (int q = 0; q < SEQ_T; ++q) {
    const float qv = qkv[base + (size_t)q * 768 + hc];
    float acc = 0.f, mmax = -1e30f, denom = 0.f;
    #pragma unroll
    for (int k = 0; k < SEQ_T; ++k) {
      float p = wave_sum(qv * kreg[k]) * scale;
      if (SOFTMAX) {
        float nm = fmaxf(mmax, p);
        float corr = __expf(mmax - nm);
        float e = __expf(p - nm);
        denom = denom * corr + e;
        acc   = acc   * corr + e * vreg[k];
        mmax = nm;
      } else {
        acc += vreg[k] / (1.0f + __expf(-p));
      }
    }
    if (SOFTMAX) acc /= denom;
    const size_t o = (size_t)(b * SEQ_T + q) * D_MODEL_N + hc;
    if (OUT_BF16) outb[o] = f32_to_bf16(acc);
    else          outf[o] = res[o] + acc;
  }
}

// ---------------- block mixing (AttentionResidual), one wave per (b,t) row ---------------
__global__ void __launch_bounds__(256)
mix_kernel(const float* __restrict__ v0, const float* __restrict__ v1,
           const float* __restrict__ v2, const float* __restrict__ v3,
           int n, const float* __restrict__ qv,
           float* __restrict__ out1, float* __restrict__ out2, int M) {
  const int wid  = blockIdx.x * 8 + (threadIdx.x >> 5);
  const int lane = threadIdx.x & 31;
  if (wid >= M) return;
  const float* Vs[4] = {v0, v1, v2, v3};

  float q8[8]; float qs = 0.f;
  #pragma unroll
  for (int j = 0; j < 8; ++j) { q8[j] = qv[lane + 32 * j]; qs += q8[j] * q8[j]; }
  qs = wave_sum(qs);
  const float qnorm = fmaxf(sqrtf(qs), 1e-12f);

  float vals[4][8];
  float logits[4];
  for (int i = 0; i < 4; ++i) {
    if (i >= n) { logits[i] = -1e30f; continue; }
    const float* vr = Vs[i] + (size_t)wid * D_MODEL_N;
    float dot = 0.f, ss = 0.f;
    #pragma unroll
    for (int j = 0; j < 8; ++j) {
      float x = vr[lane + 32 * j];
      vals[i][j] = x; dot += x * q8[j]; ss += x * x;
    }
    dot = wave_sum(dot); ss = wave_sum(ss);
    logits[i] = dot / (qnorm * fmaxf(sqrtf(ss), 1e-12f));
  }
  float mx = -1e30f;
  for (int i = 0; i < n; ++i) mx = fmaxf(mx, logits[i]);
  float wsum = 0.f; float wt[4];
  for (int i = 0; i < 4; ++i) {
    wt[i] = (i < n) ? __expf(logits[i] - mx) : 0.f;
    wsum += wt[i];
  }
  const float inv = 1.0f / wsum;
  #pragma unroll
  for (int j = 0; j < 8; ++j) {
    float o = 0.f;
    for (int i = 0; i < 4; ++i) if (i < n) o += wt[i] * vals[i][j];
    o *= inv;
    const size_t idx = (size_t)wid * D_MODEL_N + lane + 32 * j;
    out1[idx] = o;
    if (out2) out2[idx] = o;
  }
}

// ---------------- final head: out[b] = relu-hidden[b,:1024] . Wo2 + bo2 ------------------
__global__ void __launch_bounds__(256)
head_reduce_kernel(const unsigned short* __restrict__ hh, const float* __restrict__ w2,
                   const float* __restrict__ b2, float* __restrict__ out) {
  const int wid  = blockIdx.x * 8 + (threadIdx.x >> 5);   // batch index
  const int lane = threadIdx.x & 31;
  const unsigned short* hr = hh + (size_t)wid * D_FF_N;
  float s = 0.f;
  #pragma unroll
  for (int j = 0; j < 32; ++j) {
    int c = lane + 32 * j;
    s += bf16_to_f32(hr[c]) * w2[c];
  }
  s = wave_sum(s);
  if (lane == 0) out[wid] = s + b2[0];
}

// =========================================================================================
extern "C" void kernel_launch(void* const* d_in, const int* in_sizes, int n_in,
                              void* d_out, int out_size, void* d_ws, size_t ws_size,
                              hipStream_t stream) {
  (void)in_sizes; (void)n_in; (void)out_size; (void)ws_size;
  const float* dense_x  = (const float*)d_in[0];
  const int*   sparse_x = (const int*)  d_in[1];
  const float* W_dense  = (const float*)d_in[2];
  const float* b_dense  = (const float*)d_in[3];
  const float* emb      = (const float*)d_in[4];
  const float* Wq = (const float*)d_in[5];  const float* bq = (const float*)d_in[6];
  const float* Wk = (const float*)d_in[7];  const float* bk = (const float*)d_in[8];
  const float* Wv = (const float*)d_in[9];  const float* bv = (const float*)d_in[10];
  const float* W1 = (const float*)d_in[11]; const float* b1 = (const float*)d_in[12];
  const float* W2 = (const float*)d_in[13]; const float* b2 = (const float*)d_in[14];
  const float* n1g = (const float*)d_in[15]; const float* n1b = (const float*)d_in[16];
  const float* n2g = (const float*)d_in[17]; const float* n2b = (const float*)d_in[18];
  const float* qvec = (const float*)d_in[19];
  const float* mha_in_w  = (const float*)d_in[20]; const float* mha_in_b  = (const float*)d_in[21];
  const float* mha_out_w = (const float*)d_in[22]; const float* mha_out_b = (const float*)d_in[23];
  const float* ng  = (const float*)d_in[24]; const float* ngb = (const float*)d_in[25];
  const float* Wo1 = (const float*)d_in[26]; const float* bo1 = (const float*)d_in[27];
  const float* Wo2 = (const float*)d_in[28]; const float* bo2 = (const float*)d_in[29];
  float* out = (float*)d_out;

  // ---- workspace carving (256B aligned) ----
  char* ws = (char*)d_ws;
  size_t off = 0;
  auto alloc = [&](size_t bytes) { size_t r = off; off += (bytes + 255) & ~(size_t)255; return r; };
  const size_t M = M_ROWS;
  float* buf_x  = (float*)(ws + alloc(M * D_MODEL_N * 4));
  float* buf_b0 = (float*)(ws + alloc(M * D_MODEL_N * 4));
  float* buf_b1 = (float*)(ws + alloc(M * D_MODEL_N * 4));
  float* buf_b2 = (float*)(ws + alloc(M * D_MODEL_N * 4));
  unsigned short* buf_h = (unsigned short*)(ws + alloc(M * D_MODEL_N * 2));
  char* big = ws + alloc(M * 768 * 4);                 // qkv fp32 / ffn bf16 / flat bf16 (aliased)
  float*          buf_qkv  = (float*)big;
  unsigned short* buf_ffn  = (unsigned short*)big;
  unsigned short* buf_flat = (unsigned short*)big;
  unsigned short* buf_headh = (unsigned short*)(ws + alloc((size_t)BATCH_N * D_FF_N * 2));
  // pre-transposed bf16 weights: Wt[K,N]
  unsigned short* wqt = (unsigned short*)(ws + alloc((size_t)N_LAYERS_N * 256 * 256 * 2));
  unsigned short* wkt = (unsigned short*)(ws + alloc((size_t)N_LAYERS_N * 256 * 256 * 2));
  unsigned short* wvt = (unsigned short*)(ws + alloc((size_t)N_LAYERS_N * 256 * 256 * 2));
  unsigned short* w1t = (unsigned short*)(ws + alloc((size_t)N_LAYERS_N * 256 * 1024 * 2));
  unsigned short* w2t = (unsigned short*)(ws + alloc((size_t)N_LAYERS_N * 1024 * 256 * 2));
  unsigned short* mint  = (unsigned short*)(ws + alloc((size_t)256 * 768 * 2));
  unsigned short* moutt = (unsigned short*)(ws + alloc((size_t)256 * 256 * 2));
  unsigned short* wo1t  = (unsigned short*)(ws + alloc((size_t)9984 * 1024 * 2));

  auto cvtT = [&](const float* src, unsigned short* dst, int N, int K) {
    size_t n = (size_t)N * K;
    cvt_transpose_bf16_kernel<<<(unsigned)((n + 255) / 256), 256, 0, stream>>>(src, dst, N, K);
  };
  for (int l = 0; l < N_LAYERS_N; ++l) {
    cvtT(Wq + (size_t)l * 65536,  wqt + (size_t)l * 65536,  256, 256);
    cvtT(Wk + (size_t)l * 65536,  wkt + (size_t)l * 65536,  256, 256);
    cvtT(Wv + (size_t)l * 65536,  wvt + (size_t)l * 65536,  256, 256);
    cvtT(W1 + (size_t)l * 262144, w1t + (size_t)l * 262144, 1024, 256);
    cvtT(W2 + (size_t)l * 262144, w2t + (size_t)l * 262144, 256, 1024);
  }
  cvtT(mha_in_w,  mint,  768, 256);
  cvtT(mha_out_w, moutt, 256, 256);
  cvtT(Wo1,       wo1t,  1024, 9984);

  // ---- embed ----
  embed_kernel<<<(unsigned)(M * D_MODEL_N / 256), 256, 0, stream>>>(
      dense_x, sparse_x, W_dense, b_dense, emb, buf_x, buf_b0);

  const unsigned lnGrid  = (unsigned)(M / 8);                    // wave-per-row kernels
  const unsigned attGrid = (unsigned)(BATCH_N * N_HEADS_N / 8);  // wave per (b,h)
  const dim3 gD(M / 128, D_MODEL_N / 128);                       // N=256 GEMMs
  const dim3 gF1(M / 128, D_FF_N / 128);                         // N=1024 GEMM
  const float hstu_scale = 0.0625f;                              // 1/sqrt(256)
  const float mha_scale  = 0.17677669529663687f;                 // 1/sqrt(32)

  // ---- HSTU layers ----
  for (int l = 0; l < N_LAYERS_N; ++l) {
    layernorm_bf16_kernel<<<lnGrid, 256, 0, stream>>>(
        buf_x, n1g + l * 256, n1b + l * 256, buf_h, (int)M);
    // Q, K, V projections into [M,768] slices
    gemm_wmma_bf16<false, false, true, false><<<gD, 256, 0, stream>>>(
        buf_h, 256, wqt + (size_t)l * 65536, 256, 256, bq + l * 256,
        buf_qkv + 0, 768, nullptr, 0, nullptr, 0);
    gemm_wmma_bf16<false, false, true, false><<<gD, 256, 0, stream>>>(
        buf_h, 256, wkt + (size_t)l * 65536, 256, 256, bk + l * 256,
        buf_qkv + 256, 768, nullptr, 0, nullptr, 0);
    gemm_wmma_bf16<false, false, true, false><<<gD, 256, 0, stream>>>(
        buf_h, 256, wvt + (size_t)l * 65536, 256, 256, bv + l * 256,
        buf_qkv + 512, 768, nullptr, 0, nullptr, 0);
    // sigmoid pointwise attention + residual into x (in place, per-element)
    attention_kernel<false, false><<<attGrid, 256, 0, stream>>>(
        buf_qkv, hstu_scale, buf_x, buf_x, nullptr);
    layernorm_bf16_kernel<<<lnGrid, 256, 0, stream>>>(
        buf_x, n2g + l * 256, n2b + l * 256, buf_h, (int)M);
    // FFN: relu(h W1^T + b1) -> bf16 (aliased over qkv), then W2^T + b2 + residual -> x
    gemm_wmma_bf16<true, false, false, true><<<gF1, 256, 0, stream>>>(
        buf_h, 256, w1t + (size_t)l * 262144, 1024, 256, b1 + l * 1024,
        nullptr, 0, buf_ffn, 1024, nullptr, 0);
    gemm_wmma_bf16<false, true, true, false><<<gD, 256, 0, stream>>>(
        buf_ffn, 1024, w2t + (size_t)l * 262144, 256, 1024, b2 + l * 256,
        buf_x, 256, nullptr, 0, buf_x, 256);
    // block mixing after layers 1, 3, 5 (0-based)
    if (l == 1) {
      mix_kernel<<<lnGrid, 256, 0, stream>>>(buf_b0, buf_x, nullptr, nullptr, 2,
                                             qvec + 1 * 256, buf_x, buf_b1, (int)M);
    } else if (l == 3) {
      mix_kernel<<<lnGrid, 256, 0, stream>>>(buf_b0, buf_b1, buf_x, nullptr, 3,
                                             qvec + 2 * 256, buf_x, buf_b2, (int)M);
    } else if (l == 5) {
      mix_kernel<<<lnGrid, 256, 0, stream>>>(buf_b0, buf_b1, buf_b2, buf_x, 4,
                                             qvec + 3 * 256, buf_x, nullptr, (int)M);
    }
  }

  // ---- final LN + standard MHA residual ----
  layernorm_bf16_kernel<<<lnGrid, 256, 0, stream>>>(buf_x, ng, ngb, buf_h, (int)M);
  gemm_wmma_bf16<false, false, true, false><<<dim3(M / 128, 768 / 128), 256, 0, stream>>>(
      buf_h, 256, mint, 768, 256, mha_in_b, buf_qkv, 768, nullptr, 0, nullptr, 0);
  // softmax attention -> mo (bf16, reusing buf_h as next GEMM-A)
  attention_kernel<true, true><<<attGrid, 256, 0, stream>>>(
      buf_qkv, mha_scale, nullptr, nullptr, buf_h);
  // out-proj + residual -> x (fp32) and flat bf16 copy (aliased over qkv region)
  gemm_wmma_bf16<false, true, true, true><<<gD, 256, 0, stream>>>(
      buf_h, 256, moutt, 256, 256, mha_out_b, buf_x, 256, buf_flat, 256, buf_x, 256);

  // ---- output head: relu(flat Wo1^T + bo1) [4096,1024] bf16, then dot with Wo2 ----
  gemm_wmma_bf16<true, false, false, true><<<dim3(BATCH_N / 128, D_FF_N / 128), 256, 0, stream>>>(
      buf_flat, SEQ_T * D_MODEL_N, wo1t, 1024, SEQ_T * D_MODEL_N, bo1,
      nullptr, 0, buf_headh, 1024, nullptr, 0);
  head_reduce_kernel<<<BATCH_N / 8, 256, 0, stream>>>(buf_headh, Wo2, bo2, out);
}
